// RNN_87531433492488
// MI455X (gfx1250) — matrix operand order repaired and checked by reading, not compiled, not run on hardware
//
#include <hip/hip_runtime.h>

// ---------------------------------------------------------------------------
// Fused RNN on gfx1250 (MI455X): per-block batch tile of 16 rows, state h^T
// held in LDS in WMMA B-operand layout, W_ih/W_hh held in VGPRs in WMMA
// A-operand layout (one 16-wide output tile per wave, 13 waves/block).
// Recurrence: h_t = relu(W_ih * x_t^T + W_hh * h_{t-1}^T + b)  (transposed)
// All GEMM work via v_wmma_f32_16x16x32_bf16 (f32 accumulate).
//  - time loop unrolled x2: ping/pong LDS offsets are immediates
//  - all 16 B-operand ds_load_b128 issued before the 8 WMMAs (sched_barrier)
//  - x prefetch software-pipelined one full step ahead (register-carried),
//    issued unconditionally with clamped index so it cannot be sunk into a
//    predicated block
// ---------------------------------------------------------------------------

typedef __attribute__((ext_vector_type(16))) __bf16 v16bf;
typedef __attribute__((ext_vector_type(8)))  __bf16 v8bf;
typedef __attribute__((ext_vector_type(2)))  __bf16 v2bf;
typedef __attribute__((ext_vector_type(8)))  float  v8f;

#define RNN_B 8192
#define RNN_T 128
#define RNN_I 28
#define RNN_H 200
#define RNN_O 10

#define BM 16                     // batch rows per block
#define NW 13                     // waves per block: 13*16 = 208 >= H
#define NTHREADS (NW * 32)        // 416 threads
#define KC 7                      // K chunks of 32: 7*32 = 224 >= H
#define CHUNK_BYTES 1024          // one 32x16 bf16 B-operand tile

#define H_BUF_BYTES (KC * CHUNK_BYTES)              // 7168 B per h^T buffer
#define LDS_H0 0
#define LDS_H1 H_BUF_BYTES
#define LDS_X0 (2 * H_BUF_BYTES)
#define LDS_X1 (2 * H_BUF_BYTES + CHUNK_BYTES)
#define LDS_TOTAL (2 * H_BUF_BYTES + 2 * CHUNK_BYTES)  // 16384 B

#if __has_builtin(__builtin_amdgcn_sched_barrier)
#define SCHED_FENCE() __builtin_amdgcn_sched_barrier(0)
#else
#define SCHED_FENCE()
#endif

// One RNN timestep. Buffer offsets are template constants so every DS access
// is base-VGPR + immediate offset. xcarry holds x_{t+1} (loaded last step);
// on return it holds x_{t+2}.
template <int HB_CUR, int HB_NXT, int XB_CUR, int XB_NXT>
__device__ __forceinline__ void rnn_step(
    char* lds, const float* __restrict__ xrow,  // per-thread clamped x base
    const v16bf& a_ih, const v16bf (&a_hh)[KC], const v8f& cbias,
    char* hDstBase, int lane, int tid, int t, float2& xcarry)
{
  // ---- commit x_{t+1} (carried in registers; load latency already hidden).
  // thread tid<224 owns (n = tid&15, i = 2*(tid>>4), 2*(tid>>4)+1): both i's
  // are contiguous bf16 slots of one B-layout lane -> one 4B store.
  if (tid < 224) {
    int i0 = 2 * (tid >> 4);
    int l2 = (tid & 15) + 16 * (i0 >> 4);
    v2bf p;
    p[0] = (__bf16)xcarry.x;
    p[1] = (__bf16)xcarry.y;
    *(v2bf*)(lds + XB_NXT + l2 * 32 + (i0 & 15) * 2) = p;
  }

  // ---- issue prefetch of x_{t+2} (clamped, unconditional: no branch, the
  // wait lands in the NEXT step's commit, a full step of WMMAs away).
  {
    int t2 = (t + 2 < RNN_T) ? (t + 2) : (RNN_T - 1);
    xcarry = *(const float2*)(xrow + (size_t)t2 * RNN_I);
  }

  SCHED_FENCE();

  // ---- stage all 8 B operands: 16 ds_load_b128 in flight, distinct VGPRs
  v16bf bops[KC + 1];
  bops[0] = *(const v16bf*)(lds + XB_CUR + lane * 32);
#pragma unroll
  for (int kc = 0; kc < KC; ++kc)
    bops[kc + 1] =
        *(const v16bf*)(lds + HB_CUR + kc * CHUNK_BYTES + lane * 32);

  SCHED_FENCE();

  // ---- 8 back-to-back WMMAs: acc = bias + W_ih*x^T + W_hh*h^T
  v8f acc = cbias;
  acc = __builtin_amdgcn_wmma_f32_16x16x32_bf16(
      false, a_ih, false, bops[0], (short)0, acc, false, false);
#pragma unroll
  for (int kc = 0; kc < KC; ++kc)
    acc = __builtin_amdgcn_wmma_f32_16x16x32_bf16(
        false, a_hh[kc], false, bops[kc + 1], (short)0, acc, false, false);

  // ---- relu + f32->bf16 + single 16B store into next h^T buffer
  v8bf hb;
#pragma unroll
  for (int r = 0; r < 8; ++r)
    hb[r] = (__bf16)__builtin_fmaxf(acc[r], 0.f);
  *(v8bf*)(hDstBase + HB_NXT) = hb;
}

__global__ __launch_bounds__(NTHREADS) void rnn_fused_kernel(
    const float* __restrict__ x,     // [B, T, I]
    const float* __restrict__ W_ih,  // [H, I]
    const float* __restrict__ W_hh,  // [H, H]
    const float* __restrict__ b_ih,  // [H]
    const float* __restrict__ b_hh,  // [H]
    const float* __restrict__ W_fc,  // [O, H]
    const float* __restrict__ b_fc,  // [O]
    float* __restrict__ out)         // [B, O]
{
  __shared__ alignas(32) char lds[LDS_TOTAL];

  const int tid    = threadIdx.x;
  const int lane   = tid & 31;
  const int ntile  = tid >> 5;           // 0..12 : which 16-wide slice of H
  const int laneHi = (lane >> 4) & 1;
  const int mloc   = lane & 15;
  const int b0     = blockIdx.x * BM;

  // ---- zero-init LDS (padding lanes must stay zero forever) ----
  for (int i = tid; i < LDS_TOTAL / 4; i += NTHREADS)
    ((int*)lds)[i] = 0;

  // per-thread clamped x base pointer for the prefetch stream:
  // thread (tid<224) reads x[b0 + (tid&15), t, 2*(tid>>4) .. +1]
  const int xe = (tid < 224) ? tid : 0;
  const float* xrow =
      x + (size_t)(b0 + (xe & 15)) * RNN_T * RNN_I + 2 * (xe >> 4);

  // ---- gather weights from global into per-lane A-operand registers ----
  // A operand (16-bit, 16x32), contiguous k order as in B: lane = m + 16*laneHi holds, per slot s:
//   k' = s + 16*laneHi
  const int row = ntile * 16 + mloc;     // output h-unit of this lane
  v16bf a_hh[KC];
  v16bf a_ih;
#pragma unroll
  for (int kc = 0; kc < KC; ++kc) {
#pragma unroll
    for (int s = 0; s < 16; ++s) {
      int v  = s >> 1;
      int kp = s + 16 * laneHi;
      int col = kc * 32 + kp;
      float w = (row < RNN_H && col < RNN_H) ? W_hh[row * RNN_H + col] : 0.f;
      a_hh[kc][s] = (__bf16)w;
    }
  }
#pragma unroll
  for (int s = 0; s < 16; ++s) {
    int v  = s >> 1;
    int kp = s + 16 * laneHi;
    float w = (row < RNN_H && kp < RNN_I) ? W_ih[row * RNN_I + kp] : 0.f;
    a_ih[s] = (__bf16)w;
  }

  // ---- bias in C/D layout: VGPR r of lane holds M = r + 8*laneHi ----
  v8f cbias;
#pragma unroll
  for (int r = 0; r < 8; ++r) {
    int mm = ntile * 16 + r + 8 * laneHi;
    cbias[r] = (mm < RNN_H) ? (b_ih[mm] + b_hh[mm]) : 0.f;
  }

  // ---- stage x_0^T into sX buffer 0; preload x_1 into the carry regs ----
  if (tid < 224) {
    float2 x0 = *(const float2*)(xrow);  // t = 0
    int i0 = 2 * (tid >> 4);
    int l2 = (tid & 15) + 16 * (i0 >> 4);
    v2bf p;
    p[0] = (__bf16)x0.x;
    p[1] = (__bf16)x0.y;
    *(v2bf*)(lds + LDS_X0 + l2 * 32 + (i0 & 15) * 2) = p;
  }
  float2 xcarry = *(const float2*)(xrow + RNN_I);  // x_1 (all threads)
  __syncthreads();

  // D-tile -> next-step B-operand writeback coordinates (constant per lane):
  // value (r, lane) is (k' = (ntile&1)*16 + r + 8*laneHi, n = lane&15): lands
  // at chunk ntile>>1, lane' = (lane&15) + 16*(ntile&1), slots r + 8*laneHi
  // => 8 contiguous bf16 = one 16B store per lane.
  const int wlane = (lane & 15) + 16 * (ntile & 1);
  char* const hDstBase =
      lds + (ntile >> 1) * CHUNK_BYTES + wlane * 32 + laneHi * 16;

  // ---- time loop, unrolled x2 for explicit ping/pong (T=128 is even) ----
  for (int t = 0; t < RNN_T; t += 2) {
    rnn_step<LDS_H0, LDS_H1, LDS_X0, LDS_X1>(lds, xrow, a_ih, a_hh, cbias,
                                             hDstBase, lane, tid, t, xcarry);
    __syncthreads();
    rnn_step<LDS_H1, LDS_H0, LDS_X1, LDS_X0>(lds, xrow, a_ih, a_hh, cbias,
                                             hDstBase, lane, tid, t + 1,
                                             xcarry);
    __syncthreads();
  }
  // after an even number of steps, h_last lives in buffer H0

  // ---- FC head: out[b, o] = h_last . W_fc[o,:] + b_fc[o]  (33 MFLOP total)
  for (int j = tid; j < BM * RNN_O; j += NTHREADS) {
    int n = j / RNN_O, o = j % RNN_O;
    float s = b_fc[o];
#pragma unroll 8
    for (int k = 0; k < RNN_H; ++k) {
      int kc = k >> 5, kp = k & 31;
      const __bf16* hp = (const __bf16*)(lds + LDS_H0 + kc * CHUNK_BYTES +
                                         (n + 16 * (kp >> 4)) * 32 +
                                         (kp & 15) * 2);
      s += (float)(*hp) * W_fc[o * RNN_H + k];
    }
    out[(size_t)(b0 + n) * RNN_O + o] = s;
  }
}

extern "C" void kernel_launch(void* const* d_in, const int* in_sizes, int n_in,
                              void* d_out, int out_size, void* d_ws, size_t ws_size,
                              hipStream_t stream) {
  const float* x    = (const float*)d_in[0];
  const float* W_ih = (const float*)d_in[1];
  const float* W_hh = (const float*)d_in[2];
  const float* b_ih = (const float*)d_in[3];
  const float* b_hh = (const float*)d_in[4];
  const float* W_fc = (const float*)d_in[5];
  const float* b_fc = (const float*)d_in[6];
  float* out = (float*)d_out;

  dim3 grid(RNN_B / BM);          // 512 blocks
  dim3 block(NTHREADS);           // 416 threads = 13 wave32
  rnn_fused_kernel<<<grid, block, 0, stream>>>(x, W_ih, W_hh, b_ih, b_hh,
                                               W_fc, b_fc, out);
}
